// FNN_29334626632316
// MI455X (gfx1250) — matrix operand order repaired and checked by reading, not compiled, not run on hardware
//
#include <hip/hip_runtime.h>
#include <hip/hip_bf16.h>
#include <math.h>

typedef __attribute__((ext_vector_type(16))) _Float16 v16h;
typedef __attribute__((ext_vector_type(8)))  _Float16 v8h;
typedef __attribute__((ext_vector_type(8)))  float    v8f;

#define B_TOT    128
#define T_STEPS  100
#define KC       200
#define COG      64
#define TERM     16
#define BB       16      // batch elements per workgroup
#define NTHREADS 256     // 8 wave32 waves
#define KDIM     (TERM * COG)   // 1024

__global__ __launch_bounds__(NTHREADS)
void fnn_scan_kernel(const float* __restrict__ scores,   // [B, T]
                     const int*   __restrict__ skills,   // [B, T+1]
                     const float* __restrict__ mean,     // [TERM]
                     const float* __restrict__ sigma,    // [TERM]
                     const float* __restrict__ cog0,     // [KC, COG]
                     const float* __restrict__ w_cog,    // [KDIM, COG]
                     const float* __restrict__ w_pred,   // [COG]
                     const float* __restrict__ b_pred,   // [1]
                     float* __restrict__ pred_out,       // [B, T]
                     float* __restrict__ cog_out,        // [B, T, COG]
                     float* __restrict__ memg)           // [B, KC, COG] state (d_ws)
{
    // w_cog resident in LDS as f16, swizzled: wlds[((k>>4)*64 + n)*16 + (k&15)]
    // -> each lane's WMMA B-fragment is 16 contiguous halfs (32B).
    __shared__ alignas(64) _Float16 wlds[KDIM * COG];    // 128 KB
    __shared__ alignas(32) _Float16 fs_lds[BB][TERM];    // fuzzy memberships (f16)
    __shared__ alignas(32) _Float16 last_lds[BB][COG];   // gathered cognition rows (f16)
    __shared__ float cogsum[BB][COG];                    // GEMM result accumulation
    __shared__ float invsum[BB];
    __shared__ float predsum[BB];
    __shared__ int   st_lds[BB], st1_lds[BB];
    __shared__ float mean_l[TERM], sig2_l[TERM], wpred_l[COG];
    __shared__ float bpred_l;

    const int tid = threadIdx.x;
    const int blk = blockIdx.x;

    // ---- one-time staging ----
    for (int idx = tid; idx < KDIM * COG; idx += NTHREADS) {
        int k = idx >> 6, n = idx & 63;
        wlds[(((k >> 4) * 64 + n) << 4) | (k & 15)] = (_Float16)w_cog[idx];
    }
    if (tid < TERM) { mean_l[tid] = mean[tid]; float s = sigma[tid]; sig2_l[tid] = s * s; }
    if (tid < COG)  wpred_l[tid] = w_pred[tid];
    if (tid == 0)   bpred_l = b_pred[0];

    // init this block's slice of the memory state: mem[b, kc, :] = cog0[kc, :]
    for (int idx = tid; idx < BB * KC * COG; idx += NTHREADS) {
        int bb = idx / (KC * COG);
        int r  = idx - bb * (KC * COG);
        memg[(size_t)(blk * BB + bb) * (KC * COG) + r] = cog0[r];
    }
    __threadfence();
    __syncthreads();

    // ---- per-wave WMMA geometry ----
    const int wave  = tid >> 5;
    const int lane  = tid & 31;
    const int ntile = wave & 3;         // which 16-wide slice of COG
    const int khalf = wave >> 2;        // k-tiles [0,16) or [16,32)
    const int m_a   = lane & 15;        // A: row
    const int kb_a  = (lane >> 4) << 3; // A: half-group base (0 or 8)
    const int n_b   = lane & 15;        // B: column within tile
    const int kh_b  = lane >> 4;        // B: K sub-half (0 or 1)
    const int mrow  = (lane >> 4) << 3; // D: row base (0 or 8)
    const int kbase = khalf * 16;

    for (int t = 0; t < T_STEPS; ++t) {
        // zero step accumulators
        for (int idx = tid; idx < BB * COG; idx += NTHREADS) ((float*)cogsum)[idx] = 0.f;
        if (tid < BB) predsum[tid] = 0.f;

        // fuzzy memberships + skill indices (256 threads == 16x16 exactly)
        {
            int bb = tid >> 4, i = tid & 15;
            int b  = blk * BB + bb;
            float s = scores[b * T_STEPS + t];
            float d = s - mean_l[i];
            fs_lds[bb][i] = (_Float16)__expf(-d * d / sig2_l[i]);
            if (i == 0) {
                st_lds[bb]  = skills[b * (T_STEPS + 1) + t];
                st1_lds[bb] = skills[b * (T_STEPS + 1) + t + 1];
            }
        }
        __syncthreads();

        // gather last = mem[b, skill_t] -> f16
        for (int idx = tid; idx < BB * COG; idx += NTHREADS) {
            int bb = idx >> 6, j = idx & 63;
            last_lds[bb][j] = (_Float16)
                memg[(size_t)(blk * BB + bb) * (KC * COG) + st_lds[bb] * COG + j];
        }
        __syncthreads();

        // GEMM: (fs ⊗ last)[16 x 1024] @ w_cog[1024 x 64] -> cogsum[16 x 64]
        // A built on the fly in f16: rule[m][i*64+k] = fs[m][i]*last[m][k];
        // i = kt>>1 is constant per 32-wide K tile. Dual accumulators break the
        // WMMA->WMMA serial chain; full unroll exposes 16 wmma issues per wave.
        v8f acc0 = {}, acc1 = {};
        const v8h fs8 = *(const v8h*)&fs_lds[m_a][khalf * 8]; // this wave's 8 fs values
        const _Float16* lrow = &last_lds[m_a][0];
        #pragma unroll
        for (int u = 0; u < 16; ++u) {
            const int kt = kbase + u;
            const _Float16 fsv = fs8[u >> 1];     // compile-time extract
            const int ks = (u & 1) << 5;          // 0 / 32 within the COG block
            v8h lo = *(const v8h*)(lrow + ks + 2 * kb_a);
            v8h hi = *(const v8h*)(lrow + ks + 2 * kb_a + 8);
            lo = lo * fsv;                        // v_pk_mul_f16
            hi = hi * fsv;
            v16h a = __builtin_shufflevector(lo, hi,
                0, 1, 2, 3, 4, 5, 6, 7, 8, 9, 10, 11, 12, 13, 14, 15);
            const v16h bfrag = *(const v16h*)
                &wlds[(((kt * 2 + kh_b) * 64 + (ntile * 16 + n_b)) << 4)];
            if (u & 1)
                acc1 = __builtin_amdgcn_wmma_f32_16x16x32_f16(
                    false, a, false, bfrag, (short)0, acc1, false, false);
            else
                acc0 = __builtin_amdgcn_wmma_f32_16x16x32_f16(
                    false, a, false, bfrag, (short)0, acc0, false, false);
        }
        const v8f accF = acc0 + acc1;
        // combine the two K-half waves per N-tile (ds_add_f32)
        #pragma unroll
        for (int r = 0; r < 8; ++r)
            atomicAdd(&cogsum[mrow + r][ntile * 16 + n_b], accF[r]);
        __syncthreads();

        // row sums -> fast reciprocal
        if (tid < BB) {
            float s = 0.f;
            #pragma unroll 8
            for (int j = 0; j < COG; ++j) s += cogsum[tid][j];
            invsum[tid] = __builtin_amdgcn_rcpf(s);
        }
        __syncthreads();

        // normalize + scatter back into state
        for (int idx = tid; idx < BB * COG; idx += NTHREADS) {
            int bb = idx >> 6, j = idx & 63;
            float v = cogsum[bb][j] * invsum[bb];
            memg[(size_t)(blk * BB + bb) * (KC * COG) + st_lds[bb] * COG + j] = v;
        }
        __threadfence();
        __syncthreads();

        // outputs: cog_t = mem[b, skill_t1]; pred_t = clip(cog_t . w_pred + b, 0, 1)
        for (int idx = tid; idx < BB * COG; idx += NTHREADS) {
            int bb = idx >> 6, j = idx & 63;
            int b  = blk * BB + bb;
            float v = memg[(size_t)b * (KC * COG) + st1_lds[bb] * COG + j];
            cog_out[((size_t)b * T_STEPS + t) * COG + j] = v;
            atomicAdd(&predsum[bb], v * wpred_l[j]);
        }
        __syncthreads();
        if (tid < BB) {
            int b = blk * BB + tid;
            float p = predsum[tid] + bpred_l;
            pred_out[b * T_STEPS + t] = fminf(fmaxf(p, 0.f), 1.f);
        }
        __syncthreads();
    }
}

extern "C" void kernel_launch(void* const* d_in, const int* in_sizes, int n_in,
                              void* d_out, int out_size, void* d_ws, size_t ws_size,
                              hipStream_t stream) {
    const float* scores = (const float*)d_in[0];   // [128,100]
    const int*   skills = (const int*)  d_in[1];   // [128,101]
    const float* mean   = (const float*)d_in[2];   // [16]
    const float* sigma  = (const float*)d_in[3];   // [16]
    const float* cog0   = (const float*)d_in[4];   // [200,64]
    const float* w_cog  = (const float*)d_in[5];   // [1024,64]
    const float* w_pred = (const float*)d_in[6];   // [64]
    const float* b_pred = (const float*)d_in[7];   // [1]

    float* out      = (float*)d_out;
    float* pred_out = out;                           // [128,100]
    float* cog_out  = out + (size_t)B_TOT * T_STEPS; // [128,100,64]
    float* memg     = (float*)d_ws;                  // [128,200,64] state

    fnn_scan_kernel<<<dim3(B_TOT / BB), dim3(NTHREADS), 0, stream>>>(
        scores, skills, mean, sigma, cog0, w_cog, w_pred, b_pred,
        pred_out, cog_out, memg);
}
